// LSTM_36601711297132
// MI455X (gfx1250) — matrix operand order repaired and checked by reading, not compiled, not run on hardware
//
#include <hip/hip_runtime.h>

typedef __attribute__((ext_vector_type(16))) _Float16 v16h;
typedef __attribute__((ext_vector_type(8)))  float    v8f;

#define T_STEPS 1024
#define N_IN    128
#define N_HID   64
#define N_G     256      // 4*H
#define ROWS    16       // batch rows per block (one WMMA M-tile)
#define LDX     (N_IN + 2)
#define LDH     (N_HID + 2)
#define LDG     (N_G + 4)

__device__ __forceinline__ float sigm_f(float v) {
  return 1.0f / (1.0f + __expf(-v));
}
__device__ __forceinline__ float tanh_f(float v) {
  v = fminf(fmaxf(v, -15.0f), 15.0f);
  float e = __expf(2.0f * v);
  return (e - 1.0f) / (e + 1.0f);
}

__global__ __launch_bounds__(256, 1)
void bilstm_wmma_kernel(const float* __restrict__ x,
                        const float* __restrict__ Wih_f, const float* __restrict__ Whh_f,
                        const float* __restrict__ b_f,
                        const float* __restrict__ Wih_b, const float* __restrict__ Whh_b,
                        const float* __restrict__ b_b,
                        float* __restrict__ out)
{
  __shared__ _Float16 xt[ROWS][LDX];   // x tile for current t, f16
  __shared__ _Float16 hsh[ROWS][LDH];  // hidden state, f16 (A-operand of Whh GEMM)
  __shared__ float    gsh[ROWS][LDG];  // pre-activation gates [16 x 256]

  const int dir = blockIdx.x & 1;
  const int b0  = (blockIdx.x >> 1) * ROWS;
  const float* __restrict__ Wih = dir ? Wih_b : Wih_f;
  const float* __restrict__ Whh = dir ? Whh_b : Whh_f;
  const float* __restrict__ bv  = dir ? b_b   : b_f;

  const int tid   = threadIdx.x;
  const int wave  = tid >> 5;
  const int lane  = tid & 31;
  const int lcol  = lane & 15;
  const int khalf = (lane >> 4) << 4;  // B-frag: lanes 16-31 hold K+16..K+31
  const int koff  = (lane >> 4) << 4;  // A-frag: lanes 16-31 hold K+16..K+31

  // ---- weight B-fragments: loaded once, register-resident forever ----
  // gates layout: wave w owns gate columns [32w, 32w+32) as two 16-wide N-tiles
  v16h bih[2][4];   // Wih^T fragments, K-chunks of 32 (K = 128)
  v16h bhh[2][2];   // Whh^T fragments, K-chunks of 32 (K = 64)
  float bias[2];
  #pragma unroll
  for (int j = 0; j < 2; ++j) {
    const int n = 32 * wave + 16 * j + lcol;       // gate column this lane owns
    const float* wr = Wih + (size_t)n * N_IN;      // row n of Wih == column n of Wih^T
    #pragma unroll
    for (int kc = 0; kc < 4; ++kc) {
      const int kb = 32 * kc + khalf;
      v16h f;
      #pragma unroll
      for (int i = 0; i < 16; ++i) f[i] = (_Float16)wr[kb + i];
      bih[j][kc] = f;
    }
    const float* wh = Whh + (size_t)n * N_HID;
    #pragma unroll
    for (int kc = 0; kc < 2; ++kc) {
      const int kb = 32 * kc + khalf;
      v16h f;
      #pragma unroll
      for (int i = 0; i < 16; ++i) f[i] = (_Float16)wh[kb + i];
      bhh[j][kc] = f;
    }
    bias[j] = bv[32 * wave + 16 * j + lcol];
  }

  // zero initial hidden state in LDS
  for (int i = tid; i < ROWS * LDH; i += 256) ((_Float16*)hsh)[i] = (_Float16)0.0f;

  // per-thread cell state: thread owns 4 consecutive (row, col) cells for all T
  float creg[4] = {0.f, 0.f, 0.f, 0.f};
  const int erow = tid >> 4;
  const int ecol = (tid & 15) * 4;

  // x staging coords: each thread converts 8 contiguous floats of one row
  const int xr = tid >> 4;
  const int xc = (tid & 15) * 8;
  const int arow = lcol;

  for (int s = 0; s < T_STEPS; ++s) {
    const int t = dir ? (T_STEPS - 1 - s) : s;

    // ---- stage x[b0..b0+15, t, :] -> LDS f16; prefetch next step ----
    {
      const float* src = x + ((size_t)(b0 + xr) * T_STEPS + t) * N_IN + xc;
      float4 v0 = ((const float4*)src)[0];
      float4 v1 = ((const float4*)src)[1];
      xt[xr][xc + 0] = (_Float16)v0.x;  xt[xr][xc + 1] = (_Float16)v0.y;
      xt[xr][xc + 2] = (_Float16)v0.z;  xt[xr][xc + 3] = (_Float16)v0.w;
      xt[xr][xc + 4] = (_Float16)v1.x;  xt[xr][xc + 5] = (_Float16)v1.y;
      xt[xr][xc + 6] = (_Float16)v1.z;  xt[xr][xc + 7] = (_Float16)v1.w;
      if (s + 1 < T_STEPS) {
        const int tn = dir ? (t - 1) : (t + 1);
        __builtin_prefetch(x + ((size_t)(b0 + xr) * T_STEPS + tn) * N_IN + xc, 0, 0);
      }
    }
    __syncthreads();

    // ---- gates[16x32 per wave] = x_t @ Wih^T + h @ Whh^T + b ----
    v8f acc0, acc1;
    #pragma unroll
    for (int i = 0; i < 8; ++i) { acc0[i] = bias[0]; acc1[i] = bias[1]; }

    #pragma unroll
    for (int kc = 0; kc < 4; ++kc) {
      v16h a;
      const _Float16* p = &xt[arow][kc * 32 + koff];
      #pragma unroll
      for (int i = 0; i < 8; ++i) a[i] = p[i];          // K = kc*32 + koff + 0..7
      #pragma unroll
      for (int i = 0; i < 8; ++i) a[8 + i] = p[8 + i]; // K = kc*32 + koff + 8..15
      acc0 = __builtin_amdgcn_wmma_f32_16x16x32_f16(false, a, false, bih[0][kc],
                                                    (short)0, acc0, false, false);
      acc1 = __builtin_amdgcn_wmma_f32_16x16x32_f16(false, a, false, bih[1][kc],
                                                    (short)0, acc1, false, false);
    }
    #pragma unroll
    for (int kc = 0; kc < 2; ++kc) {
      v16h a;
      const _Float16* p = &hsh[arow][kc * 32 + koff];
      #pragma unroll
      for (int i = 0; i < 8; ++i) a[i] = p[i];
      #pragma unroll
      for (int i = 0; i < 8; ++i) a[8 + i] = p[8 + i];
      acc0 = __builtin_amdgcn_wmma_f32_16x16x32_f16(false, a, false, bhh[0][kc],
                                                    (short)0, acc0, false, false);
      acc1 = __builtin_amdgcn_wmma_f32_16x16x32_f16(false, a, false, bhh[1][kc],
                                                    (short)0, acc1, false, false);
    }

    // scatter accumulators (C layout: row = v + 8*(lane>=16), col = lane%16)
    {
      const int rb = (lane < 16) ? 0 : 8;
      #pragma unroll
      for (int v = 0; v < 8; ++v) {
        gsh[rb + v][32 * wave + lcol]      = acc0[v];
        gsh[rb + v][32 * wave + 16 + lcol] = acc1[v];
      }
    }
    __syncthreads();

    // ---- LSTM cell update: gate order i,f,g,o at column offsets 0/64/128/192 ----
    float hv[4];
    #pragma unroll
    for (int q = 0; q < 4; ++q) {
      const float gi = gsh[erow][ecol + q];
      const float gf = gsh[erow][64 + ecol + q];
      const float gg = gsh[erow][128 + ecol + q];
      const float go = gsh[erow][192 + ecol + q];
      const float cn = sigm_f(gf) * creg[q] + sigm_f(gi) * tanh_f(gg);
      creg[q] = cn;
      const float hn = sigm_f(go) * tanh_f(cn);
      hv[q] = hn;
      hsh[erow][ecol + q] = (_Float16)hn;   // feeds next step's Whh GEMM
    }
    // out[b, t, dir*64 + col], out is [B, T, 2H]
    float* op = out + ((size_t)(b0 + erow) * T_STEPS + t) * (2 * N_HID)
                    + dir * N_HID + ecol;
    *(float4*)op = *(const float4*)hv;
    // no trailing barrier needed: next iteration's first __syncthreads()
    // separates this step's hsh writes from the next step's hsh reads.
  }
}

extern "C" void kernel_launch(void* const* d_in, const int* in_sizes, int n_in,
                              void* d_out, int out_size, void* d_ws, size_t ws_size,
                              hipStream_t stream) {
  (void)in_sizes; (void)n_in; (void)out_size; (void)d_ws; (void)ws_size;
  const float* x     = (const float*)d_in[0];
  const float* Wih_f = (const float*)d_in[1];
  const float* Whh_f = (const float*)d_in[2];
  const float* b_f   = (const float*)d_in[3];
  const float* Wih_b = (const float*)d_in[4];
  const float* Whh_b = (const float*)d_in[5];
  const float* b_b   = (const float*)d_in[6];
  float* out = (float*)d_out;

  dim3 grid((512 / ROWS) * 2);  // 32 batch-chunks x 2 directions = 64 blocks
  dim3 block(256);              // 8 wave32s
  bilstm_wmma_kernel<<<grid, block, 0, stream>>>(x, Wih_f, Whh_f, b_f,
                                                 Wih_b, Whh_b, b_b, out);
}
